// DenseGeneral_70892730188041
// MI455X (gfx1250) — 8-bit matrix operand order repaired and checked by reading, compiled, not run on hardware
//
#include <hip/hip_runtime.h>
#include <hip/hip_bf16.h>

typedef __attribute__((ext_vector_type(16))) int   v16i;
typedef __attribute__((ext_vector_type(8)))  float v8f;

#define E4M3_MAX 448.0f
#define D_DIM 512
#define F_DIM 512
#define M_TILE 64

__device__ __forceinline__ float qclamp(float x, float inv) {
    float y = x * inv;
    return fminf(fmaxf(y, -E4M3_MAX), E4M3_MAX);
}

__device__ __forceinline__ unsigned pack4_e4m3(float a, float b, float c, float d) {
    unsigned v = (unsigned)__builtin_amdgcn_cvt_pk_fp8_f32(a, b, 0, false);
    v = (unsigned)__builtin_amdgcn_cvt_pk_fp8_f32(c, d, (int)v, true);
    return v;
}

__global__ __launch_bounds__(256) void quantB_kernel(const float* __restrict__ W,
                                                     const float* __restrict__ kscale,
                                                     unsigned* __restrict__ wsB) {
    int id = blockIdx.x * 256 + threadIdx.x;
    int n  = id >> 7;
    int k4 = id & 127;
    float inv = 1.0f / kscale[0];
    float a = qclamp(W[(size_t)(k4 * 4 + 0) * F_DIM + n], inv);
    float b = qclamp(W[(size_t)(k4 * 4 + 1) * F_DIM + n], inv);
    float c = qclamp(W[(size_t)(k4 * 4 + 2) * F_DIM + n], inv);
    float d = qclamp(W[(size_t)(k4 * 4 + 3) * F_DIM + n], inv);
    wsB[(size_t)n * 128 + k4] = pack4_e4m3(a, b, c, d);
}

__global__ __launch_bounds__(512) void fp8_gemm_kernel(const float* __restrict__ x,
                                                       const unsigned* __restrict__ wsB,
                                                       const float* __restrict__ bias,
                                                       const float* __restrict__ iscale,
                                                       const float* __restrict__ kscale,
                                                       float* __restrict__ out) {
    __shared__ unsigned ldsA[M_TILE * (D_DIM / 4)];

    const int t  = threadIdx.x;
    const int m0 = blockIdx.x * M_TILE;

    {
        const float inv = 1.0f / iscale[0];
#pragma unroll
        for (int i = 0; i < 16; ++i) {
            int idx  = i * 512 + t;
            int row  = idx >> 7;
            int c4   = idx & 127;
            const float4 v = *(const float4*)(x + (size_t)(m0 + row) * D_DIM + c4 * 4);
            ldsA[row * 128 + c4] = pack4_e4m3(qclamp(v.x, inv), qclamp(v.y, inv),
                                              qclamp(v.z, inv), qclamp(v.w, inv));
        }
    }
    __syncthreads();

    const int lane = t & 31;
    const int w    = t >> 5;
    const int mw   = w & 3;
    const int nw   = (w >> 2) * 128;
    const int lcol = lane & 15;
    const int lhi  = lane >> 4;

    v8f acc[8];
#pragma unroll
    for (int n = 0; n < 8; ++n)
#pragma unroll
        for (int v = 0; v < 8; ++v) acc[n][v] = 0.0f;

    const char* ldsBytes = (const char*)ldsA;

#pragma unroll
    for (int kc = 0; kc < D_DIM; kc += 128) {
        v16i a;
#pragma unroll
        for (int blk = 0; blk < 2; ++blk) {
#pragma unroll
            for (int kg = 0; kg < 4; ++kg) {
                uint2 dq = *(const uint2*)(ldsBytes + (mw * 16 + lcol) * D_DIM
                                           + kc + blk * 64 + (kg >> 1) * 32 + lhi * 16 + (kg & 1) * 8);
                int e = blk * 8 + kg * 2;
                a[e]     = (int)dq.x;
                a[e + 1] = (int)dq.y;
            }
        }

#pragma unroll
        for (int n = 0; n < 8; ++n) {
            const char* bp = (const char*)wsB
                             + (size_t)(nw + n * 16 + lcol) * D_DIM + kc + lhi * 16;
            v16i b;
#pragma unroll
            for (int g = 0; g < 4; ++g) {
                uint4 q = *(const uint4*)(bp + g * 32);
                b[4 * g + 0] = (int)q.x;
                b[4 * g + 1] = (int)q.y;
                b[4 * g + 2] = (int)q.z;
                b[4 * g + 3] = (int)q.w;
            }
            acc[n] = __builtin_amdgcn_wmma_f32_16x16x128_fp8_fp8(
                a, b, (short)0, acc[n], false, false);
        }
    }

    const float s = iscale[0] * kscale[0];
#pragma unroll
    for (int n = 0; n < 8; ++n) {
        int col = nw + n * 16 + lcol;
        float bb = __bfloat162float(__float2bfloat16(bias[col]));
#pragma unroll
        for (int v = 0; v < 8; ++v) {
            int row = m0 + mw * 16 + v + 8 * lhi;
            out[(size_t)row * F_DIM + col] = acc[n][v] * s + bb;
        }
    }
}

extern "C" void kernel_launch(void* const* d_in, const int* in_sizes, int n_in,
                              void* d_out, int out_size, void* d_ws, size_t ws_size,
                              hipStream_t stream) {
    const float* x      = (const float*)d_in[0];
    const float* W      = (const float*)d_in[1];
    const float* bias   = (const float*)d_in[2];
    const float* iscale = (const float*)d_in[3];
    const float* kscale = (const float*)d_in[4];
    float*       out    = (float*)d_out;
    unsigned*    wsB    = (unsigned*)d_ws;

    const int M = in_sizes[0] / D_DIM;

    quantB_kernel<<<(F_DIM * (D_DIM / 4)) / 256, 256, 0, stream>>>(W, kscale, wsB);

    fp8_gemm_kernel<<<M / M_TILE, 512, 0, stream>>>(x, wsB, bias, iscale, kscale, out);
}
